// ROLLOUT_10445360463953
// MI455X (gfx1250) — matrix operand order repaired and checked by reading, not compiled, not run on hardware
//
#include <hip/hip_runtime.h>
#include <hip/hip_bf16.h>
#include <stdint.h>

// Problem constants (from reference)
#define BB   64
#define TT   32
#define VV   10000
#define EE   256
#define HH   512
#define VPAD 10048   // V padded to multiple of 64 (16-col tiles x 4-way N blocking)

typedef __bf16 bf16_t;
typedef __attribute__((ext_vector_type(8)))  bf16_t v8bf;
typedef __attribute__((ext_vector_type(16))) bf16_t v16bf;
typedef __attribute__((ext_vector_type(8)))  float  v8f;

// ---------------------------------------------------------------------------
// WMMA fragment loaders (ISA 7.12.2 layouts, wave32)
// ---------------------------------------------------------------------------
__device__ __forceinline__ v16bf join16(v8bf lo, v8bf hi) {
  return __builtin_shufflevector(lo, hi, 0,1,2,3,4,5,6,7,8,9,10,11,12,13,14,15);
}

// A-matrix 16x32 bf16, source row-major [M x K] with leading dim ld.
// lane L: row m = mBase + (L&15); k runs: K0+kb..+7 and K0+kb+16..+23, kb = (L>=16)?8:0
__device__ __forceinline__ v16bf load_a_frag(const bf16_t* A, int ld, int mBase, int K0) {
  int lane = threadIdx.x & 31;
  const bf16_t* p = A + (mBase + (lane & 15)) * ld + K0 + ((lane >> 4) << 3);
  v8bf lo = *(const v8bf*)(p);
  v8bf hi = *(const v8bf*)(p + 16);
  return join16(lo, hi);
}

// B-matrix 32x16 bf16 from PRE-TRANSPOSED weights Bt row-major [N x K], ld = K.
// lane L: col n = nBase + (L&15); k runs: K0+kb..+7 and K0+kb+16..+23, kb = (L>=16)?8:0
__device__ __forceinline__ v16bf load_b_frag(const bf16_t* Bt, int ld, int nBase, int K0) {
  int lane = threadIdx.x & 31;
  const bf16_t* p = Bt + (nBase + (lane & 15)) * ld + K0 + ((lane >> 4) << 3);
  v8bf lo = *(const v8bf*)(p);
  v8bf hi = *(const v8bf*)(p + 16);
  return join16(lo, hi);
}

__device__ __forceinline__ float sigmoidf_(float x) {
  return 1.0f / (1.0f + __expf(-x));
}

// ---------------------------------------------------------------------------
// Weight convert + transpose: in f32 [K x N] -> out bf16 [Npad x K] (n>=N -> 0)
// ---------------------------------------------------------------------------
__global__ void convT_kernel(const float* __restrict__ in, bf16_t* __restrict__ out,
                             int K, int N, int Npad) {
  long idx = (long)blockIdx.x * blockDim.x + threadIdx.x;
  long total = (long)Npad * K;
  if (idx >= total) return;
  int n = (int)(idx / K);
  int k = (int)(idx % K);
  out[idx] = (n < N) ? (bf16_t)in[(long)k * N + n] : (bf16_t)0.0f;
}

// ---------------------------------------------------------------------------
// Init: h0 = 0 (bf16), c0 = 0 (f32), x0 = embed(start_token) (bf16)
// ---------------------------------------------------------------------------
__global__ void init_kernel(const int* __restrict__ start_token,
                            const float* __restrict__ g_emb,
                            bf16_t* __restrict__ xb, bf16_t* __restrict__ hb,
                            float* __restrict__ c) {
  int idx = blockIdx.x * blockDim.x + threadIdx.x;   // 0 .. B*H-1
  if (idx < BB * HH) { hb[idx] = (bf16_t)0.0f; c[idx] = 0.0f; }
  if (idx < BB * EE) {
    int b = idx / EE, e = idx % EE;
    xb[idx] = (bf16_t)g_emb[(long)start_token[b] * EE + e];
  }
}

// ---------------------------------------------------------------------------
// Fused LSTM cell: one wave = one 16x16 (b,h) tile, 4 gate accumulators.
// pre[g] = x @ W[g] + h @ U[g] + bias[g]; then cell update in-register.
// grid: 32 blocks x 128 threads (4 waves) -> 4 mTiles * 32 nTiles = 128 tiles.
// ---------------------------------------------------------------------------
__global__ __launch_bounds__(128)
void gates_kernel(const bf16_t* __restrict__ xb,     // [B,E] bf16
                  const bf16_t* __restrict__ hb,     // [B,H] bf16 (current)
                  const float*  __restrict__ c_in,   // [B,H] f32 (current)
                  const bf16_t* __restrict__ WT,     // [4][H,E] bf16 (transposed)
                  const bf16_t* __restrict__ UT,     // [4][H,H] bf16 (transposed)
                  const float* __restrict__ bi, const float* __restrict__ bfg,
                  const float* __restrict__ bog, const float* __restrict__ bc,
                  bf16_t* __restrict__ hb_out,       // [B,H] bf16 (next)
                  float*  __restrict__ c_out) {      // [B,H] f32 (next)
  int wave  = threadIdx.x >> 5;
  int tile  = blockIdx.x * 4 + wave;      // 0..127
  int mBase = (tile >> 5) * 16;           // 4 m tiles (B=64)
  int nBase = (tile & 31) * 16;           // 32 n tiles (H=512)

  v8f acc[4] = {v8f{}, v8f{}, v8f{}, v8f{}};

  // x @ W (K = E)
  for (int k0 = 0; k0 < EE; k0 += 32) {
    v16bf a = load_a_frag(xb, EE, mBase, k0);
    #pragma unroll
    for (int g = 0; g < 4; ++g) {
      v16bf bm = load_b_frag(WT + (long)g * HH * EE, EE, nBase, k0);
      acc[g] = __builtin_amdgcn_wmma_f32_16x16x32_bf16(false, a, false, bm,
                                                       (short)0, acc[g], false, false);
    }
  }
  // h @ U (K = H)
  for (int k0 = 0; k0 < HH; k0 += 32) {
    v16bf a = load_a_frag(hb, HH, mBase, k0);
    #pragma unroll
    for (int g = 0; g < 4; ++g) {
      v16bf bm = load_b_frag(UT + (long)g * HH * HH, HH, nBase, k0);
      acc[g] = __builtin_amdgcn_wmma_f32_16x16x32_bf16(false, a, false, bm,
                                                       (short)0, acc[g], false, false);
    }
  }

  // Cell update. C/D layout: lane L -> col n = nBase+(L&15); VGPR r -> row
  // m = mBase + 8*(L>>4) + r.
  int lane = threadIdx.x & 31;
  int n    = nBase + (lane & 15);
  int mOff = (lane >> 4) * 8;
  float vbi = bi[n], vbf = bfg[n], vbo = bog[n], vbc = bc[n];
  #pragma unroll
  for (int r = 0; r < 8; ++r) {
    int b   = mBase + mOff + r;
    float i = sigmoidf_(acc[0][r] + vbi);
    float f = sigmoidf_(acc[1][r] + vbf);
    float o = sigmoidf_(acc[2][r] + vbo);
    float cb = tanhf(acc[3][r] + vbc);
    float cn = f * c_in[b * HH + n] + i * cb;
    c_out[b * HH + n]  = cn;
    hb_out[b * HH + n] = (bf16_t)(o * tanhf(cn));
  }
}

// ---------------------------------------------------------------------------
// Logits GEMM: logits[B,VPAD] = h[B,H] @ WoT[VPAD,H]^T + bo.
// 4-way N blocking per wave: one wave computes a 16x64 slab with 4
// accumulators sharing each A fragment (4 WMMAs per A load).
// grid: (ceil(157/4)=40, 4 mTiles) x 128 threads; wave guard is uniform.
// ---------------------------------------------------------------------------
__global__ __launch_bounds__(128)
void logits_kernel(const bf16_t* __restrict__ hb,   // [B,H]
                   const bf16_t* __restrict__ WoT,  // [VPAD,H]
                   const float*  __restrict__ bo,   // [V]
                   float* __restrict__ logits) {    // [B,VPAD]
  int wave  = threadIdx.x >> 5;
  int nBase = (blockIdx.x * 4 + wave) * 64;
  if (nBase >= VPAD) return;               // uniform per wave
  int mBase = blockIdx.y * 16;

  v8f acc[4] = {v8f{}, v8f{}, v8f{}, v8f{}};
  for (int k0 = 0; k0 < HH; k0 += 32) {
    v16bf a = load_a_frag(hb, HH, mBase, k0);
    #pragma unroll
    for (int j = 0; j < 4; ++j) {
      v16bf bm = load_b_frag(WoT, HH, nBase + j * 16, k0);
      acc[j] = __builtin_amdgcn_wmma_f32_16x16x32_bf16(false, a, false, bm,
                                                       (short)0, acc[j], false, false);
    }
  }

  int lane = threadIdx.x & 31;
  int nOff = lane & 15;
  int mOff = (lane >> 4) * 8;
  #pragma unroll
  for (int j = 0; j < 4; ++j) {
    int n = nBase + j * 16 + nOff;
    float bias = (n < VV) ? bo[n] : 0.0f;
    #pragma unroll
    for (int r = 0; r < 8; ++r)
      logits[(long)(mBase + mOff + r) * VPAD + n] = acc[j][r] + bias;
  }
}

// ---------------------------------------------------------------------------
// Gumbel-max categorical sampling + teacher forcing. One block per batch row.
// ---------------------------------------------------------------------------
__device__ __forceinline__ uint64_t splitmix64(uint64_t z) {
  z += 0x9E3779B97F4A7C15ull;
  z = (z ^ (z >> 30)) * 0xBF58476D1CE4E5B9ull;
  z = (z ^ (z >> 27)) * 0x94D049BB133111EBull;
  return z ^ (z >> 31);
}

__global__ __launch_bounds__(256)
void sample_kernel(const float* __restrict__ logits,   // [B,VPAD]
                   const int* __restrict__ input_x,    // [B,T]
                   const int* __restrict__ given_num,  // scalar
                   int t, int* __restrict__ out_tok) { // [B,T]
  int b = blockIdx.x;
  __shared__ float sval[256];
  __shared__ int   sidx[256];
  float best = -INFINITY; int besti = 0;
  for (int v = threadIdx.x; v < VV; v += 256) {
    uint64_t key = ((uint64_t)(uint32_t)(t * BB + b) << 32) ^ (uint32_t)v;
    uint64_t r = splitmix64(key ^ 0x243F6A8885A308D3ull);
    float u = ((float)(uint32_t)(r >> 40) + 0.5f) * (1.0f / 16777216.0f); // (0,1)
    float g = -__logf(-__logf(u));
    float val = logits[(long)b * VPAD + v] + g;
    if (val > best) { best = val; besti = v; }
  }
  sval[threadIdx.x] = best; sidx[threadIdx.x] = besti;
  __syncthreads();
  for (int s = 128; s > 0; s >>= 1) {
    if (threadIdx.x < s && sval[threadIdx.x + s] > sval[threadIdx.x]) {
      sval[threadIdx.x] = sval[threadIdx.x + s];
      sidx[threadIdx.x] = sidx[threadIdx.x + s];
    }
    __syncthreads();
  }
  if (threadIdx.x == 0) {
    int tok = (t < *given_num) ? input_x[b * TT + t] : sidx[0];
    out_tok[b * TT + t] = tok;
  }
}

// ---------------------------------------------------------------------------
// Embedding gather for next step input (to bf16)
// ---------------------------------------------------------------------------
__global__ __launch_bounds__(256)
void embed_kernel(const int* __restrict__ out_tok, int t,
                  const float* __restrict__ g_emb, bf16_t* __restrict__ xb) {
  int idx = blockIdx.x * blockDim.x + threadIdx.x;   // 0 .. B*E-1
  int b = idx / EE, e = idx % EE;
  int tok = out_tok[b * TT + t];
  xb[idx] = (bf16_t)g_emb[(long)tok * EE + e];
}

// ---------------------------------------------------------------------------
// Host launcher
// ---------------------------------------------------------------------------
static inline size_t align256(size_t x) { return (x + 255) & ~(size_t)255; }

extern "C" void kernel_launch(void* const* d_in, const int* in_sizes, int n_in,
                              void* d_out, int out_size, void* d_ws, size_t ws_size,
                              hipStream_t stream) {
  // Inputs (setup_inputs order)
  const int*   input_x     = (const int*)  d_in[0];
  const int*   given_num   = (const int*)  d_in[1];
  const int*   start_token = (const int*)  d_in[2];
  const float* g_emb       = (const float*)d_in[3];
  const float* Wi  = (const float*)d_in[4];
  const float* Ui  = (const float*)d_in[5];
  const float* bi  = (const float*)d_in[6];
  const float* Wf  = (const float*)d_in[7];
  const float* Uf  = (const float*)d_in[8];
  const float* bfg = (const float*)d_in[9];
  const float* Wog = (const float*)d_in[10];
  const float* Uog = (const float*)d_in[11];
  const float* bog = (const float*)d_in[12];
  const float* Wc  = (const float*)d_in[13];
  const float* Uc  = (const float*)d_in[14];
  const float* bc  = (const float*)d_in[15];
  const float* Wo  = (const float*)d_in[16];
  const float* bo  = (const float*)d_in[17];

  int* out_tok = (int*)d_out;   // [B,T] int32 tokens

  // Workspace carve-up
  char* p = (char*)d_ws;
  bf16_t* WT  = (bf16_t*)p; p += align256((size_t)4 * HH * EE * sizeof(bf16_t));
  bf16_t* UT  = (bf16_t*)p; p += align256((size_t)4 * HH * HH * sizeof(bf16_t));
  bf16_t* WoT = (bf16_t*)p; p += align256((size_t)VPAD * HH * sizeof(bf16_t));
  bf16_t* xb  = (bf16_t*)p; p += align256((size_t)BB * EE * sizeof(bf16_t));
  bf16_t* hb0 = (bf16_t*)p; p += align256((size_t)BB * HH * sizeof(bf16_t));
  bf16_t* hb1 = (bf16_t*)p; p += align256((size_t)BB * HH * sizeof(bf16_t));
  float*  c0  = (float*)p;  p += align256((size_t)BB * HH * sizeof(float));
  float*  c1  = (float*)p;  p += align256((size_t)BB * HH * sizeof(float));
  float*  logits = (float*)p; p += align256((size_t)BB * VPAD * sizeof(float));
  (void)ws_size; (void)n_in; (void)in_sizes; (void)out_size;

  // --- Weight convert + transpose (bf16, [N x K] layout for B fragments) ---
  {
    long tW = (long)HH * EE;                       // per W matrix
    int  gW = (int)((tW + 255) / 256);
    convT_kernel<<<gW, 256, 0, stream>>>(Wi,  WT + 0L * HH * EE, EE, HH, HH);
    convT_kernel<<<gW, 256, 0, stream>>>(Wf,  WT + 1L * HH * EE, EE, HH, HH);
    convT_kernel<<<gW, 256, 0, stream>>>(Wog, WT + 2L * HH * EE, EE, HH, HH);
    convT_kernel<<<gW, 256, 0, stream>>>(Wc,  WT + 3L * HH * EE, EE, HH, HH);
    long tU = (long)HH * HH;
    int  gU = (int)((tU + 255) / 256);
    convT_kernel<<<gU, 256, 0, stream>>>(Ui,  UT + 0L * HH * HH, HH, HH, HH);
    convT_kernel<<<gU, 256, 0, stream>>>(Uf,  UT + 1L * HH * HH, HH, HH, HH);
    convT_kernel<<<gU, 256, 0, stream>>>(Uog, UT + 2L * HH * HH, HH, HH, HH);
    convT_kernel<<<gU, 256, 0, stream>>>(Uc,  UT + 3L * HH * HH, HH, HH, HH);
    long tO = (long)VPAD * HH;
    int  gO = (int)((tO + 255) / 256);
    convT_kernel<<<gO, 256, 0, stream>>>(Wo, WoT, HH, VV, VPAD);
  }

  // --- Initial state ---
  init_kernel<<<(BB * HH + 255) / 256, 256, 0, stream>>>(start_token, g_emb, xb, hb0, c0);

  bf16_t* h_cur = hb0; bf16_t* h_nxt = hb1;
  float*  c_cur = c0;  float*  c_nxt = c1;

  const int nSlabs      = VPAD / 64;               // 157 16x64 slabs
  const int logitsGridX = (nSlabs + 3) / 4;        // 40
  for (int t = 0; t < TT; ++t) {
    gates_kernel<<<32, 128, 0, stream>>>(xb, h_cur, c_cur, WT, UT,
                                         bi, bfg, bog, bc, h_nxt, c_nxt);
    logits_kernel<<<dim3(logitsGridX, BB / 16), 128, 0, stream>>>(h_nxt, WoT, bo, logits);
    sample_kernel<<<BB, 256, 0, stream>>>(logits, input_x, given_num, t, out_tok);
    if (t + 1 < TT)
      embed_kernel<<<(BB * EE + 255) / 256, 256, 0, stream>>>(out_tok, t, g_emb, xb);
    // swap double buffers
    bf16_t* ht = h_cur; h_cur = h_nxt; h_nxt = ht;
    float*  ct = c_cur; c_cur = c_nxt; c_nxt = ct;
  }
}
